// HMM_4157528342776
// MI455X (gfx1250) — matrix operand order repaired and checked by reading, not compiled, not run on hardware
//
#include <hip/hip_runtime.h>

#define NVOCAB 32000
#define NSTATE 512
#define NBATCH 64
#define MAXLEN 256
#define ROW_CHUNKS 32
#define ROWS_PER_CHUNK (NVOCAB / ROW_CHUNKS)

typedef __attribute__((ext_vector_type(16))) _Float16 v16h;
typedef __attribute__((ext_vector_type(8)))  _Float16 v8h;
typedef __attribute__((ext_vector_type(8)))  float    v8f;

union V16U { v16h v; v8h h[2]; };

// ---------- small helpers ----------
__device__ __forceinline__ void lse_acc(float x, float& m, float& s) {
  if (x > m) { s = s * __expf(m - x) + 1.0f; m = x; }
  else       { s += __expf(x - m); }
}
__device__ __forceinline__ void lse_merge(float& m, float& s, float m2, float s2) {
  float M = fmaxf(m, m2);
  s = s * __expf(m - M) + s2 * __expf(m2 - M);
  m = M;
}
__device__ float block_reduce_max(float v, float* red) {
  int t = threadIdx.x;
  red[t] = v; __syncthreads();
  for (int off = 128; off; off >>= 1) {
    if (t < off) red[t] = fmaxf(red[t], red[t + off]);
    __syncthreads();
  }
  float r = red[0]; __syncthreads();
  return r;
}
__device__ float block_reduce_sum(float v, float* red) {
  int t = threadIdx.x;
  red[t] = v; __syncthreads();
  for (int off = 128; off; off >>= 1) {
    if (t < off) red[t] += red[t + off];
    __syncthreads();
  }
  float r = red[0]; __syncthreads();
  return r;
}

// ---------- kernel 1: partial column logsumexp of input_w [VOCAB, S] over vocab ----------
__global__ void col_lse_part(const float* __restrict__ w, float* __restrict__ pm,
                             float* __restrict__ ps) {
  int col = blockIdx.x * 256 + threadIdx.x;      // 2 blocks in x cover 512 cols
  int chunk = blockIdx.y;
  int r0 = chunk * ROWS_PER_CHUNK;
  float m = -3.4e38f, s = 0.0f;
  for (int r = 0; r < ROWS_PER_CHUNK; ++r) {
    float x = w[(size_t)(r0 + r) * NSTATE + col];
    lse_acc(x, m, s);
  }
  pm[chunk * NSTATE + col] = m;
  ps[chunk * NSTATE + col] = s;
}

// ---------- kernel 2: merge partials -> lse_col[512] ----------
__global__ void col_lse_final(const float* __restrict__ pm, const float* __restrict__ ps,
                              float* __restrict__ lse_col) {
  int col = threadIdx.x;                          // 512 threads, 1 block
  float M = -3.4e38f;
  for (int c = 0; c < ROW_CHUNKS; ++c) M = fmaxf(M, pm[c * NSTATE + col]);
  float Ssum = 0.0f;
  for (int c = 0; c < ROW_CHUNKS; ++c) Ssum += ps[c * NSTATE + col] * __expf(pm[c * NSTATE + col] - M);
  lse_col[col] = M + __logf(Ssum);
}

// ---------- kernel 3: transition prep ----------
// T = log_softmax(transition, axis=-1).  expT'[i][j] = exp(x[i][j]-rowmax_i) = exp(T[i][j]-Tmax_i),
// tmax[i] = Tmax_i = -log(sum_j exp(x[i][j]-rowmax_i)).
// expT' stored f16 in WMMA B-fragment layout: tile (nt over i, kt over j):
//   lane = (i&15) + 16*((j&31)>>4), element e = j&15  ->  lanes 0-15: K 0..15, lanes 16-31: K 16..31.
__global__ void trans_prep(const float* __restrict__ trans, _Float16* __restrict__ expTB,
                           float* __restrict__ tmax) {
  __shared__ float red[256];
  int i = blockIdx.x, t = threadIdx.x;
  float x0 = trans[i * NSTATE + t], x1 = trans[i * NSTATE + t + 256];
  float xmax = block_reduce_max(fmaxf(x0, x1), red);
  float e0 = __expf(x0 - xmax), e1 = __expf(x1 - xmax);
  float Ssum = block_reduce_sum(e0 + e1, red);
  if (t == 0) tmax[i] = -__logf(Ssum);
  int nt = i >> 4, ll = i & 15;
  for (int q = 0; q < 2; ++q) {
    int j = t + q * 256;
    float ev = q ? e1 : e0;
    int kt = j >> 5, jj = j & 31;
    int lane = ll + ((jj >> 4) << 4);
    int e = jj & 15;
    expTB[(size_t)((((nt << 4) + kt) << 5) + lane) * 16 + e] = (_Float16)ev;
  }
}

// ---------- kernel 4: m0[i] = lse_j(x[i,j]+begin[j]) - lseRow_i - lse(begin) ----------
__global__ void init_m0(const float* __restrict__ trans, const float* __restrict__ begin,
                        float* __restrict__ m0) {
  __shared__ float red[256];
  int i = blockIdx.x, t = threadIdx.x;
  float b0 = begin[t], b1 = begin[t + 256];
  float bmax = block_reduce_max(fmaxf(b0, b1), red);
  float bS = block_reduce_sum(__expf(b0 - bmax) + __expf(b1 - bmax), red);
  float lseBeg = bmax + __logf(bS);
  float x0 = trans[i * NSTATE + t], x1 = trans[i * NSTATE + t + 256];
  float xmax = block_reduce_max(fmaxf(x0, x1), red);
  float xS = block_reduce_sum(__expf(x0 - xmax) + __expf(x1 - xmax), red);
  float lseRow = xmax + __logf(xS);
  float y0 = x0 + b0, y1 = x1 + b1;
  float ymax = block_reduce_max(fmaxf(y0, y1), red);
  float yS = block_reduce_sum(__expf(y0 - ymax) + __expf(y1 - ymax), red);
  float lseY = ymax + __logf(yS);
  if (t == 0) m0[i] = lseY - lseRow - lseBeg;
}

// ---------- kernel 5: persistent single-WGP scan ----------
// LDS (dynamic, 320KB exactly): bufA[64][512] f32, bufB[64][512] f32, Pbuf[64][512] f16.
// Per step t: PhaseA: out += mask * lse_s(m_t + emis_t)  (reads cur)
//             PhaseB: P = exp(log_softmax_batch(m_{t-1}+emis_t))  (reads prev, writes Pbuf)
//             PhaseC: m_{t+1}[b,i] = tmax_i + log( sum_j expT'[i,j] * P[b,j] )   via WMMA f16
__global__ __launch_bounds__(1024) void hmm_scan(
    const int* __restrict__ sent, const float* __restrict__ maskv,
    const float* __restrict__ input_w, const float* __restrict__ lse_col,
    const float* __restrict__ tmaxg, const float* __restrict__ m0g,
    const _Float16* __restrict__ expTB, float* __restrict__ out) {
  extern __shared__ char smem[];
  float*    bufA = (float*)smem;                      // 131072 B
  float*    bufB = (float*)(smem + 131072);           // 131072 B
  _Float16* Pbuf = (_Float16*)(smem + 262144);        //  65536 B

  const int tid = threadIdx.x;
  const int w = tid >> 5;           // wave 0..31  (also the GEMM N-tile)
  const int l = tid & 31;
  const int ll = l & 15, half = l >> 4;

  // broadcast m0 across batch
  for (int k = tid; k < NBATCH * NSTATE; k += 1024) bufA[k] = m0g[k & (NSTATE - 1)];
  const float rtmax = tmaxg[(w << 4) + ll];
  float outAcc = 0.0f;

  float* prev = bufA;
  float* cur  = bufA;
  __syncthreads();

  for (int t = 0; t < MAXLEN; ++t) {
    // ---- Phase A: output contribution with cur = m_t ----
    {
      int b = (w << 1) | half;
      int v = sent[b * MAXLEN + t];
      const float* iw = input_w + (size_t)v * NSTATE;
      float m = -3.4e38f, ssum = 0.0f;
      for (int s = ll; s < NSTATE; s += 16)
        lse_acc(cur[b * NSTATE + s] + iw[s] - lse_col[s], m, ssum);
      for (int mk = 1; mk <= 8; mk <<= 1) {
        float m2 = __shfl_xor(m, mk, 32);
        float s2 = __shfl_xor(ssum, mk, 32);
        lse_merge(m, ssum, m2, s2);
      }
      if (ll == 0) outAcc += maskv[b * MAXLEN + t] * (m + __logf(ssum));
    }

    if (t < MAXLEN - 1) {
      float* nxt = (cur == bufA) ? bufB : bufA;   // overwrites prev's storage (dead after Phase B)

      // ---- Phase B: cross-batch log-softmax of z = m_{t-1}+emis_t; write P = exp(.) ----
      {
        int s  = tid >> 1;
        int b0 = (tid & 1) << 5;
        float zc = lse_col[s];
        float m = -3.4e38f, ssum = 0.0f;
        for (int bb = 0; bb < 32; ++bb) {
          int b = b0 + bb;
          int v = sent[b * MAXLEN + t];
          lse_acc(prev[b * NSTATE + s] + input_w[(size_t)v * NSTATE + s] - zc, m, ssum);
        }
        float m2 = __shfl_xor(m, 1, 32);
        float s2 = __shfl_xor(ssum, 1, 32);
        lse_merge(m, ssum, m2, s2);
        float lseB = m + __logf(ssum);
        for (int bb = 0; bb < 32; ++bb) {
          int b = b0 + bb;
          int v = sent[b * MAXLEN + t];
          float z = prev[b * NSTATE + s] + input_w[(size_t)v * NSTATE + s] - zc;
          Pbuf[b * NSTATE + s] = (_Float16)__expf(z - lseB);
        }
      }
      __syncthreads();

      // ---- Phase C: WMMA GEMM  C[b,i] = sum_j P[b,j] * expT'[i,j], N-tile = wave id ----
      {
        v8f acc[4] = {};
        for (int kt = 0; kt < 16; ++kt) {
          const _Float16* gb = expTB + (size_t)((((w << 4) + kt) << 5) + l) * 16;
          V16U bf;
          bf.h[0] = *(const v8h*)gb;
          bf.h[1] = *(const v8h*)(gb + 8);
          // WGP-scope prefetch of next K-tile's B fragment (locality=3 -> near caches);
          // SYS-scope would stop at GL2 where
          // expTB is already resident.
          if (kt < 15) __builtin_prefetch((const void*)(gb + 512), 0, 3);
          int k0 = (kt << 5) + (half << 4);        // A lanes 0-15: K 0..15; 16-31: K 16..31
          #pragma unroll
          for (int mt = 0; mt < 4; ++mt) {
            const _Float16* pa = Pbuf + ((mt << 4) + ll) * NSTATE + k0;
            V16U af;
            af.h[0] = *(const v8h*)pa;
            af.h[1] = *(const v8h*)(pa + 8);
            acc[mt] = __builtin_amdgcn_wmma_f32_16x16x32_f16(
                false, af.v, false, bf.v, (short)0, acc[mt], false, false);
          }
        }
        // write m_{t+1}: C VGPR r <-> M = r (lanes 0-15) / 8+r (lanes 16-31); N = lane&15
        int ibase = (w << 4) + ll;
        int bh = half << 3;
        #pragma unroll
        for (int mt = 0; mt < 4; ++mt) {
          #pragma unroll
          for (int r = 0; r < 8; ++r) {
            int b = (mt << 4) + bh + r;
            nxt[b * NSTATE + ibase] = rtmax + __logf(fmaxf(acc[mt][r], 1e-37f));
          }
        }
      }
      __syncthreads();
      prev = cur; cur = nxt;
    }
  }

  // final reduction of outAcc over 1024 threads
  for (int mk = 16; mk >= 1; mk >>= 1) outAcc += __shfl_xor(outAcc, mk, 32);
  __syncthreads();
  if (l == 0) bufA[w] = outAcc;
  __syncthreads();
  if (tid == 0) {
    float s = 0.0f;
    for (int i = 0; i < 32; ++i) s += bufA[i];
    out[0] = s;
  }
}

extern "C" void kernel_launch(void* const* d_in, const int* in_sizes, int n_in,
                              void* d_out, int out_size, void* d_ws, size_t ws_size,
                              hipStream_t stream) {
  const int*   sent  = (const int*)d_in[0];
  const float* maskv = (const float*)d_in[1];
  const float* iw    = (const float*)d_in[2];
  const float* trans = (const float*)d_in[3];
  const float* begin = (const float*)d_in[4];

  char* ws = (char*)d_ws;
  float*    lse_col = (float*)(ws + 0);          //  2 KB
  float*    tmax    = (float*)(ws + 2048);       //  2 KB
  float*    m0      = (float*)(ws + 4096);       //  2 KB
  _Float16* expTB   = (_Float16*)(ws + 8192);    // 512 KB
  float*    pm      = (float*)(ws + 8192 + 524288);           // 64 KB
  float*    ps      = (float*)(ws + 8192 + 524288 + 65536);   // 64 KB

  col_lse_part<<<dim3(2, ROW_CHUNKS), 256, 0, stream>>>(iw, pm, ps);
  col_lse_final<<<1, 512, 0, stream>>>(pm, ps, lse_col);
  trans_prep<<<512, 256, 0, stream>>>(trans, expTB, tmax);
  init_m0<<<512, 256, 0, stream>>>(trans, begin, m0);
  hmm_scan<<<1, 1024, 327680, stream>>>(sent, maskv, iw, lse_col, tmax, m0, expTB,
                                        (float*)d_out);
}
